// BiModalAttention_27728308863161
// MI455X (gfx1250) — matrix operand order repaired and checked by reading, not compiled, not run on hardware
//
#include <hip/hip_runtime.h>

// Problem constants (from reference): S=1024, B=32, D=1024, fp32 in/out.
#define S_LEN 1024
#define B_N   32
#define D_LEN 1024
#define BM    64                      // rows per workgroup block
#define P_STRIDE 2048                 // bytes per P row (1024 bf16)
#define P_BYTES  (BM * P_STRIDE)      // 128 KB
#define KS_OFF   P_BYTES              // K staging buffer 0
#define KS_SIZE  65536                // 64 KB per staging buffer
#define LDS_BYTES (P_BYTES + 2 * KS_SIZE)  // 256 KB total

typedef __attribute__((ext_vector_type(16))) __bf16 v16bf;
typedef __attribute__((ext_vector_type(8)))  float  v8f;

union Op16 { uint4 u[2]; v16bf v; };

// gfx1250 async copy: global -> LDS, 16 bytes per lane, tracked by ASYNCcnt.
__device__ __forceinline__ void async_cp16(unsigned lds_addr, const void* gaddr) {
  asm volatile("global_load_async_to_lds_b128 %0, %1, off"
               :: "v"(lds_addr), "v"(gaddr) : "memory");
}

// ---------------- Pass 0a: fp32 [S,B,D] -> bf16 row-major [mode,b,s,d] ----------------
__global__ __launch_bounds__(256) void k_convert(const float* __restrict__ in0,
                                                 const float* __restrict__ in1,
                                                 __bf16* __restrict__ Wrm) {
  const size_t idx = (size_t)blockIdx.x * 256 + threadIdx.x;  // 2^24 total, 4 elems each
  const int d4 = (int)(idx & 255);
  const int s  = (int)((idx >> 8) & 1023);
  const int bb = (int)((idx >> 18) & 31);
  const int m  = (int)(idx >> 23);
  const float* src = m ? in1 : in0;
  const float4 v = *(const float4*)(src + ((size_t)s * B_N + bb) * D_LEN + d4 * 4);
  __bf16* dst = Wrm + (((size_t)(m * B_N + bb) * S_LEN + s) * D_LEN + d4 * 4);
  dst[0] = (__bf16)v.x; dst[1] = (__bf16)v.y; dst[2] = (__bf16)v.z; dst[3] = (__bf16)v.w;
}

// ---------------- Pass 0b: fp32 [S,B,D] -> bf16 transposed [mode,b,d,s] ----------------
__global__ __launch_bounds__(256) void k_transpose(const float* __restrict__ in0,
                                                   const float* __restrict__ in1,
                                                   __bf16* __restrict__ Wt) {
  __shared__ float tile[32][33];
  const int tx = threadIdx.x, ty = threadIdx.y;          // 32 x 8
  const int dt = blockIdx.x * 32, st = blockIdx.y * 32;
  const int mb = blockIdx.z;
  const int m = mb >> 5, bb = mb & 31;
  const float* src = m ? in1 : in0;
#pragma unroll
  for (int j = 0; j < 4; ++j) {
    const int s = st + ty + j * 8;
    tile[ty + j * 8][tx] = src[((size_t)s * B_N + bb) * D_LEN + dt + tx];
  }
  __syncthreads();
#pragma unroll
  for (int j = 0; j < 4; ++j) {
    const int d = dt + ty + j * 8;
    Wt[((size_t)(m * B_N + bb) * D_LEN + d) * S_LEN + st + tx] = (__bf16)tile[tx][ty + j * 8];
  }
}

// ---------------- Fused attention: scores -> softmax -> PV -> gate -> writeout ----------------
__global__ __launch_bounds__(512) void k_attn(const float* __restrict__ in0,
                                              const float* __restrict__ in1,
                                              const __bf16* __restrict__ Wrm,
                                              const __bf16* __restrict__ Wt,
                                              float* __restrict__ out) {
  extern __shared__ char smem[];
  const int tid = threadIdx.x;
  const int wv = tid >> 5, lane = tid & 31, hf = lane >> 4, l16 = lane & 15;
  const int mblk = blockIdx.x, b = blockIdx.y, dir = blockIdx.z;
  const int qm = dir, km = 1 - dir;
  const int m0 = mblk * BM;

  const __bf16* Qrm = Wrm + ((size_t)(qm * B_N + b)) * S_LEN * D_LEN;
  const __bf16* Krm = Wrm + ((size_t)(km * B_N + b)) * S_LEN * D_LEN;
  const __bf16* Vt  = Wt  + ((size_t)(km * B_N + b)) * D_LEN * S_LEN;

  // ---------- Phase 1: scores[16 x 256] per wave (wave = (m-tile, key-quarter)) ----------
  const int mt = wv & 3, kq = wv >> 2;
  const int rowm = mt * 16;
  v8f acc[16];                        // 128 VGPRs of accumulators
#pragma unroll
  for (int i = 0; i < 16; ++i) acc[i] = (v8f)0.0f;

  // Per-thread staging addresses: LDS offsets are dc-invariant; global ptrs step 32 elems.
  const unsigned sbase = (unsigned)(uintptr_t)smem;   // low 32 bits = LDS byte offset
  unsigned soff[8];
  const __bf16* gsrc[8];
  {
    const int c = tid & 3;
#pragma unroll
    for (int j = 0; j < 8; ++j) {
      const int key = j * 128 + (tid >> 2);
      soff[j] = sbase + KS_OFF + key * 64 + ((c ^ ((key >> 2) & 3)) << 4);
      gsrc[j] = Krm + (size_t)key * D_LEN + c * 8;
    }
  }
  // Prologue: async-stage slice 0 into buffer 0 (K[1024 keys x 32 d] bf16, swizzled).
#pragma unroll
  for (int j = 0; j < 8; ++j) async_cp16(soff[j], gsrc[j]);

  for (int dc = 0; dc < 32; ++dc) {
    const int d0 = dc * 32;
    const unsigned cur = (dc & 1) ? KS_SIZE : 0;
    if (dc < 31) {
      // Prefetch next slice into the other buffer; it completes while we compute.
      const unsigned nxt = ((dc + 1) & 1) ? KS_SIZE : 0;
#pragma unroll
      for (int j = 0; j < 8; ++j) async_cp16(soff[j] + nxt, gsrc[j] + d0 + 32);
      // Async loads complete in order: <=8 outstanding means current slice is done.
      asm volatile("s_wait_asynccnt 0x8" ::: "memory");
    } else {
      asm volatile("s_wait_asynccnt 0x0" ::: "memory");
    }
    __syncthreads();   // publish current buffer to all waves

    // A operand (Q): lanes 0-15 hold K=d0+{0..7,16..23}; lanes 16-31 K=d0+{8..15,24..31}
    const __bf16* qp = Qrm + (size_t)(m0 + rowm + l16) * D_LEN + d0 + hf * 8;
    Op16 ao;
    ao.u[0] = *(const uint4*)qp;
    ao.u[1] = *(const uint4*)(qp + 16);
    const char* ksbuf = smem + KS_OFF + cur;
#pragma unroll
    for (int nt = 0; nt < 16; ++nt) {
      // B operand (K^T): lane col = key, lanes 0-15 K=d0+{0..7,16..23}, lanes 16-31 K=d0+{8..15,24..31}
      const int key = kq * 256 + nt * 16 + l16;
      const char* kr = ksbuf + key * 64;
      const int sw = (key >> 2) & 3;
      Op16 bo;
      bo.u[0] = *(const uint4*)(kr + (((hf)     ^ sw) << 4));
      bo.u[1] = *(const uint4*)(kr + (((hf + 2) ^ sw) << 4));
      acc[nt] = __builtin_amdgcn_wmma_f32_16x16x32_bf16(false, ao.v, false, bo.v,
                                                        (short)0, acc[nt], false, false);
    }
    __syncthreads();   // retire readers before this buffer is overwritten
  }

  // ---------- Softmax (fp32 scores in registers; cross-quarter via LDS stats) ----------
  float* smax = (float*)(smem + KS_OFF);   // [64][4]
  float* ssum = smax + 256;                // [64][4]
  float rmax[8];
#pragma unroll
  for (int r = 0; r < 8; ++r) {
    float m = acc[0][r];
#pragma unroll
    for (int nt = 1; nt < 16; ++nt) m = fmaxf(m, acc[nt][r]);
    m = fmaxf(m, __shfl_xor(m, 1, 32));
    m = fmaxf(m, __shfl_xor(m, 2, 32));
    m = fmaxf(m, __shfl_xor(m, 4, 32));
    m = fmaxf(m, __shfl_xor(m, 8, 32));
    rmax[r] = m;
  }
  if (l16 == 0) {
#pragma unroll
    for (int r = 0; r < 8; ++r) smax[(rowm + hf * 8 + r) * 4 + kq] = rmax[r];
  }
  __syncthreads();
  float rsum[8];
#pragma unroll
  for (int r = 0; r < 8; ++r) {
    const int row = rowm + hf * 8 + r;
    const float g = fmaxf(fmaxf(smax[row * 4 + 0], smax[row * 4 + 1]),
                          fmaxf(smax[row * 4 + 2], smax[row * 4 + 3]));
    float s = 0.f;
#pragma unroll
    for (int nt = 0; nt < 16; ++nt) {
      const float e = __expf(acc[nt][r] - g);
      acc[nt][r] = e;
      s += e;
    }
    s += __shfl_xor(s, 1, 32);
    s += __shfl_xor(s, 2, 32);
    s += __shfl_xor(s, 4, 32);
    s += __shfl_xor(s, 8, 32);
    rsum[r] = s;
  }
  if (l16 == 0) {
#pragma unroll
    for (int r = 0; r < 8; ++r) ssum[(rowm + hf * 8 + r) * 4 + kq] = rsum[r];
  }
  __syncthreads();

  // ---------- Normalize and store P (bf16) into swizzled LDS tile ----------
#pragma unroll
  for (int r = 0; r < 8; ++r) {
    const int row = rowm + hf * 8 + r;
    const float inv = 1.0f / (ssum[row * 4 + 0] + ssum[row * 4 + 1] +
                              ssum[row * 4 + 2] + ssum[row * 4 + 3]);
#pragma unroll
    for (int nt = 0; nt < 16; ++nt) {
      const int col = kq * 256 + nt * 16 + l16;
      const __bf16 h = (__bf16)(acc[nt][r] * inv);
      *(__bf16*)(smem + row * P_STRIDE +
                 (((col >> 3) ^ (row & 15)) << 4) + ((col & 7) << 1)) = h;
    }
  }
  __syncthreads();

  // ---------- Phase 2: O[64 x 64] per wave (wave owns dv slab) ----------
  const int dvb = wv * 64;
  v8f o[4][4];                        // 128 VGPRs of accumulators
#pragma unroll
  for (int i = 0; i < 4; ++i)
#pragma unroll
    for (int j = 0; j < 4; ++j) o[i][j] = (v8f)0.0f;

  for (int kt = 0; kt < 32; ++kt) {
    const int kb = kt * 32;
    v16bf bv[4];
#pragma unroll
    for (int n = 0; n < 4; ++n) {
      // B operand (V): lane col = dv, K = keys kb+{8h..8h+7, 16+8h..16+8h+7} (transposed bf16 copy)
      const __bf16* vp = Vt + (size_t)(dvb + n * 16 + l16) * S_LEN + kb + hf * 8;
      Op16 t;
      t.u[0] = *(const uint4*)vp;
      t.u[1] = *(const uint4*)(vp + 16);
      bv[n] = t.v;
    }
#pragma unroll
    for (int mt2 = 0; mt2 < 4; ++mt2) {
      const int row = mt2 * 16 + l16;
      const char* pr = smem + row * P_STRIDE;
      const int ch = (kb >> 3) + hf;
      const int sw = row & 15;
      Op16 ao2;
      ao2.u[0] = *(const uint4*)(pr + ((ch ^ sw) << 4));
      ao2.u[1] = *(const uint4*)(pr + (((ch + 2) ^ sw) << 4));
#pragma unroll
      for (int n = 0; n < 4; ++n)
        o[mt2][n] = __builtin_amdgcn_wmma_f32_16x16x32_bf16(false, ao2.v, false, bv[n],
                                                            (short)0, o[mt2][n], false, false);
    }
  }

  // ---------- Epilogue: gate with original fp32 Q, write [S,B,2D] ----------
  const float* qsrc = dir ? in1 : in0;
#pragma unroll
  for (int mt2 = 0; mt2 < 4; ++mt2) {
#pragma unroll
    for (int r = 0; r < 8; ++r) {
      const int srow = m0 + mt2 * 16 + hf * 8 + r;
      const size_t base_in  = ((size_t)srow * B_N + b) * D_LEN;
      const size_t base_out = ((size_t)srow * B_N + b) * (2 * D_LEN) + (size_t)dir * D_LEN;
#pragma unroll
      for (int n = 0; n < 4; ++n) {
        const int dv = dvb + n * 16 + l16;
        out[base_out + dv] = o[mt2][n][r] * qsrc[base_in + dv];
      }
    }
  }
}

extern "C" void kernel_launch(void* const* d_in, const int* in_sizes, int n_in,
                              void* d_out, int out_size, void* d_ws, size_t ws_size,
                              hipStream_t stream) {
  (void)in_sizes; (void)n_in; (void)out_size; (void)ws_size;
  const float* in0 = (const float*)d_in[0];   // mode1 fp32 [S,B,D]
  const float* in1 = (const float*)d_in[1];   // mode2 fp32 [S,B,D]
  float* out = (float*)d_out;                 // fp32 [S,B,2D]

  // Workspace: 128 MB row-major bf16 + 128 MB transposed bf16.
  __bf16* Wrm = (__bf16*)d_ws;
  __bf16* Wt  = Wrm + (size_t)2 * B_N * S_LEN * D_LEN;

  k_convert<<<dim3(65536), dim3(256), 0, stream>>>(in0, in1, Wrm);
  k_transpose<<<dim3(32, 32, 64), dim3(32, 8), 0, stream>>>(in0, in1, Wt);
  k_attn<<<dim3(16, 32, 2), dim3(512), LDS_BYTES, stream>>>(in0, in1, Wrm, Wt, out);
}
